// ResNet_88347477279406
// MI455X (gfx1250) — compile-verified, matrix operand order checked by reading, not run on hardware
//
#include <hip/hip_runtime.h>
#include <hip/hip_bf16.h>

typedef __bf16 bf16;
typedef __attribute__((ext_vector_type(16))) __bf16 v16bf;
typedef __attribute__((ext_vector_type(8)))  float  v8f;
typedef __attribute__((ext_vector_type(4)))  unsigned int u32x4;

// ---------------------------------------------------------------------------
// Implicit-GEMM conv via V_WMMA_F32_16X16X32_BF16, wave32.
// A (MxK) rows = output pixels (NHWC, M = N*Ho*Wo); B (Kx16) cols = out chans;
// weights stored [Cout][K] with K=(kh*KW+kw)*Cin+ci so both fragments load as
// two contiguous 16B chunks per lane (ISA 7.12.2 16-bit layout: lane L covers
// row/col L&15; element e -> K-offset 8*(L>>4)+e, +16 extra for e>=8).
//
// Block = 64 threads = 2 waves. Block tile = 64(M) x 64(Cout):
//   wave w -> co half  cbase = blockIdx.y*64 + w*32, sub-tiles +0 and +16
//   each wave holds 8 accumulators: 4 M-subtiles (mbase+{0,16,32,48}) x 2 co
// => per 32-wide K step: 4 A-frag loads + 2 B-frag loads feed 8 WMMAs
//    (1.5 x b128 loads per WMMA); the A-frags are shared by both waves (L0).
// M and Cout are multiples of 64 for every layer.
// ---------------------------------------------------------------------------
__global__ __launch_bounds__(64) void conv_wmma(
    const bf16* __restrict__ in, const bf16* __restrict__ wt,
    float* __restrict__ out,
    int H, int W, int Cin, int Cout,
    int stride, int pad, int KH, int KW,
    int wo_shift, int howo_shift)
{
  const int lane  = threadIdx.x & 31;
  const int wave  = threadIdx.x >> 5;
  const int mbase = blockIdx.x * 64;
  const int cbase = blockIdx.y * 64 + wave * 32;
  const int kbase = (lane >> 4) * 8;
  const int K     = KH * KW * Cin;
  int co[2];
  co[0] = cbase + (lane & 15);
  co[1] = co[0] + 16;

  // Per-M-subtile row decode (rows identical across the 2 waves).
  int nimg[4], ih0[4], iw0[4];
  #pragma unroll
  for (int t = 0; t < 4; ++t) {
    const int r   = mbase + t * 16 + (lane & 15);
    nimg[t]       = r >> howo_shift;
    const int pix = r & ((1 << howo_shift) - 1);
    const int oh  = pix >> wo_shift;
    const int ow  = pix & ((1 << wo_shift) - 1);
    ih0[t] = oh * stride - pad;
    iw0[t] = ow * stride - pad;
  }

  v8f acc[8] = {};  // acc[t*2+u]: M-subtile t, co-subtile u

  if ((Cin & 31) == 0) {
    // Fast path: every 32-wide K-step lies inside one (kh,kw) slice.
    for (int kh = 0; kh < KH; ++kh) {
      for (int kw = 0; kw < KW; ++kw) {
        const int sbase = (kh * KW + kw) * Cin + kbase;
        const bf16* bp0 = wt + (size_t)co[0] * K + sbase;
        const bf16* bp1 = wt + (size_t)co[1] * K + sbase;
        const bf16* ap[4];
        bool ok[4];
        #pragma unroll
        for (int t = 0; t < 4; ++t) {
          const int ih = ih0[t] + kh, iw = iw0[t] + kw;
          ok[t] = (ih >= 0) & (ih < H) & (iw >= 0) & (iw < W);
          ap[t] = in + ((size_t)(nimg[t] * H + ih) * W + iw) * Cin + kbase;
        }
        for (int ci = 0; ci < Cin; ci += 32) {
          union { v16bf v; u32x4 u[2]; } B0, B1;
          B0.u[0] = *(const u32x4*)(bp0 + ci);
          B0.u[1] = *(const u32x4*)(bp0 + ci + 16);
          B1.u[0] = *(const u32x4*)(bp1 + ci);
          B1.u[1] = *(const u32x4*)(bp1 + ci + 16);
          #pragma unroll
          for (int t = 0; t < 4; ++t) {
            v16bf av = {};
            if (ok[t]) {
              union { v16bf v; u32x4 u[2]; } A;
              A.u[0] = *(const u32x4*)(ap[t] + ci);
              A.u[1] = *(const u32x4*)(ap[t] + ci + 16);
              av = A.v;
            }
            acc[t * 2 + 0] = __builtin_amdgcn_wmma_f32_16x16x32_bf16(
                false, av, false, B0.v, (short)0, acc[t * 2 + 0], false, false);
            acc[t * 2 + 1] = __builtin_amdgcn_wmma_f32_16x16x32_bf16(
                false, av, false, B1.v, (short)0, acc[t * 2 + 1], false, false);
          }
        }
      }
    }
  } else {
    // Generic gather path (conv1: Cin=3, K=27 -> one padded K step).
    for (int k0 = 0; k0 < K; k0 += 32) {
      v16bf b[2] = {{}, {}};
      #pragma unroll
      for (int u = 0; u < 2; ++u)
        for (int e = 0; e < 16; ++e) {
          const int koff = (e < 8) ? (kbase + e) : (16 + kbase + (e - 8));
          const int k = k0 + koff;
          b[u][e] = (k < K) ? wt[(size_t)co[u] * K + k] : (bf16)0.0f;
        }
      #pragma unroll
      for (int t = 0; t < 4; ++t) {
        v16bf a = {};
        for (int e = 0; e < 16; ++e) {
          const int koff = (e < 8) ? (kbase + e) : (16 + kbase + (e - 8));
          const int k = k0 + koff;
          bf16 av = (bf16)0.0f;
          if (k < K) {
            const int ci = k % Cin;
            const int v  = k / Cin;
            const int kw = v % KW;
            const int kh = v / KW;
            const int ih = ih0[t] + kh, iw = iw0[t] + kw;
            if (ih >= 0 && ih < H && iw >= 0 && iw < W)
              av = in[((size_t)(nimg[t] * H + ih) * W + iw) * Cin + ci];
          }
          a[e] = av;
        }
        acc[t * 2 + 0] = __builtin_amdgcn_wmma_f32_16x16x32_bf16(
            false, a, false, b[0], (short)0, acc[t * 2 + 0], false, false);
        acc[t * 2 + 1] = __builtin_amdgcn_wmma_f32_16x16x32_bf16(
            false, a, false, b[1], (short)0, acc[t * 2 + 1], false, false);
      }
    }
  }

  // C/D layout: lanes 0-15 -> M = base+j, lanes 16-31 -> M = base+8+j.
  #pragma unroll
  for (int t = 0; t < 4; ++t) {
    const int mrow = mbase + t * 16 + ((lane >> 4) * 8);
    #pragma unroll
    for (int u = 0; u < 2; ++u)
      #pragma unroll
      for (int j = 0; j < 8; ++j)
        out[(size_t)(mrow + j) * Cout + co[u]] = acc[t * 2 + u][j];
  }
}

// ---------------------------------------------------------------------------
// Elementwise / reduction helpers
// ---------------------------------------------------------------------------
__global__ void zero_f32(float* p, int n) {
  int i = blockIdx.x * blockDim.x + threadIdx.x;
  if (i < n) p[i] = 0.f;
}

// Per-channel sum / sum^2 over M rows of an [M][C] f32 tensor.
__global__ void bn_stats(const float* __restrict__ x, float* __restrict__ sum,
                         float* __restrict__ sumsq, int M, int C) {
  int c = blockIdx.x * blockDim.x + threadIdx.x;
  if (c >= C) return;
  float s = 0.f, q = 0.f;
  for (int m = blockIdx.y; m < M; m += gridDim.y) {
    float v = x[(size_t)m * C + c];
    s += v; q += v * v;
  }
  atomicAdd(&sum[c], s);
  atomicAdd(&sumsq[c], q);
}

// y = bn(x)[*g+b] (+relu) (+resid) -> bf16.  C is a power of two (cmask=C-1).
__global__ void bn_apply(const float* __restrict__ x, const float* __restrict__ sum,
                         const float* __restrict__ sumsq,
                         const float* __restrict__ g, const float* __restrict__ bta,
                         const bf16* __restrict__ resid, bf16* __restrict__ out,
                         int n, int cmask, int M, int do_relu, int do_add) {
  int i = blockIdx.x * blockDim.x + threadIdx.x;
  if (i >= n) return;
  int c = i & cmask;
  float invM = 1.f / (float)M;
  float mean = sum[c] * invM;
  float var  = sumsq[c] * invM - mean * mean;
  float rstd = rsqrtf(var + 1e-5f);
  float y = (x[i] - mean) * rstd * g[c] + bta[c];
  if (do_relu) y = fmaxf(y, 0.f);
  if (do_add)  y += (float)resid[i];
  out[i] = (bf16)y;
}

__global__ void relu_bf16_k(const bf16* __restrict__ in, bf16* __restrict__ out, int n) {
  int i = blockIdx.x * blockDim.x + threadIdx.x;
  if (i < n) out[i] = (bf16)fmaxf((float)in[i], 0.f);
}

// OIHW f32 -> [O][(kh*KW+kw)*Cin+ci] bf16
__global__ void prep_w(const float* __restrict__ w, bf16* __restrict__ o,
                       int O, int Cin, int KH, int KW) {
  int i = blockIdx.x * blockDim.x + threadIdx.x;
  int K = Cin * KH * KW;
  if (i >= O * K) return;
  int oo = i / K, k = i % K;
  int ci = k % Cin;
  int t  = k / Cin;
  int kw = t % KW;
  int kh = t / KW;
  o[i] = (bf16)w[((size_t)(oo * Cin + ci) * KH + kh) * KW + kw];
}

// NCHW f32 -> NHWC bf16
__global__ void nchw_to_nhwc(const float* __restrict__ x, bf16* __restrict__ o,
                             int C, int HW, int total) {
  int i = blockIdx.x * blockDim.x + threadIdx.x;
  if (i >= total) return;
  int p = i % HW;
  int t = i / HW;
  int c = t % C;
  int n = t / C;
  o[((size_t)n * HW + p) * C + c] = (bf16)x[i];
}

// Global 4x4 avg-pool + FC(512->10). feat: [N*16][512] bf16 NHWC.
__global__ __launch_bounds__(128) void pool_fc(const bf16* __restrict__ feat,
                                               const float* __restrict__ Wfc,
                                               const float* __restrict__ bias,
                                               float* __restrict__ out) {
  __shared__ float f[512];
  int n = blockIdx.x;
  int t = threadIdx.x;
  for (int c = t; c < 512; c += 128) {
    float s = 0.f;
    for (int p = 0; p < 16; ++p) s += (float)feat[((size_t)n * 16 + p) * 512 + c];
    f[c] = s * (1.f / 16.f);
  }
  __syncthreads();
  if (t < 10) {
    float s = bias[t];
    for (int c = 0; c < 512; ++c) s += f[c] * Wfc[t * 512 + c];
    out[n * 10 + t] = s;
  }
}

// ---------------------------------------------------------------------------
// Host orchestration
// ---------------------------------------------------------------------------
extern "C" void kernel_launch(void* const* d_in, const int* in_sizes, int n_in,
                              void* d_out, int out_size, void* d_ws, size_t ws_size,
                              hipStream_t stream) {
  (void)in_sizes; (void)n_in; (void)out_size; (void)ws_size;
  const int NB = 256;

  // ---- parse inputs (setup_inputs dict insertion order) ----
  int ii = 0;
  const float* x_in = (const float*)d_in[ii++];
  const float* w0   = (const float*)d_in[ii++];
  const float* g0   = (const float*)d_in[ii++];
  const float* b0   = (const float*)d_in[ii++];

  const int chans[8]   = {64, 64, 128, 128, 256, 256, 512, 512};
  const int cins[8]    = {64, 64, 64, 128, 128, 256, 256, 512};
  const int strides[8] = {1, 1, 2, 1, 2, 1, 2, 1};
  const int hins[8]    = {32, 32, 32, 16, 16, 8, 8, 4};
  const bool hassc[8]  = {false, false, true, false, true, false, true, false};

  struct Blk { const float *w1,*w2,*g1,*b1,*g2,*b2,*ws,*gs,*bs; } blk[8];
  for (int i = 0; i < 8; ++i) {
    ii++;  // 'stride' python scalar (known statically)
    blk[i].w1 = (const float*)d_in[ii++];
    blk[i].w2 = (const float*)d_in[ii++];
    blk[i].g1 = (const float*)d_in[ii++];
    blk[i].b1 = (const float*)d_in[ii++];
    blk[i].g2 = (const float*)d_in[ii++];
    blk[i].b2 = (const float*)d_in[ii++];
    blk[i].ws = blk[i].gs = blk[i].bs = nullptr;
    if (hassc[i]) {
      blk[i].ws = (const float*)d_in[ii++];
      blk[i].gs = (const float*)d_in[ii++];
      blk[i].bs = (const float*)d_in[ii++];
    }
  }
  const float* Wfc  = (const float*)d_in[ii++];
  const float* bias = (const float*)d_in[ii++];

  // ---- workspace carve-up ----
  char* wsb = (char*)d_ws;
  size_t off = 0;
  auto alloc = [&](size_t bytes) { size_t o = off; off += (bytes + 255) & ~(size_t)255; return o; };
  const size_t MAXE = (size_t)NB * 32 * 32 * 64;  // 16.7M elems (largest tensor)
  bf16*  actA  = (bf16*)(wsb + alloc(MAXE * 2));
  bf16*  actB  = (bf16*)(wsb + alloc(MAXE * 2));
  bf16*  actS  = (bf16*)(wsb + alloc(MAXE * 2));
  float* convF = (float*)(wsb + alloc(MAXE * 4));
  float* stats = (float*)(wsb + alloc(4096));      // sum[512] + sumsq[512]

  auto wbuf = [&](int O, int I, int KH, int KW) {
    return (bf16*)(wsb + alloc((size_t)O * I * KH * KW * 2));
  };
  bf16* wt0 = wbuf(64, 3, 3, 3);
  bf16 *wt1[8], *wt2[8], *wtS[8];
  for (int i = 0; i < 8; ++i) {
    wt1[i] = wbuf(chans[i], cins[i], 3, 3);
    wt2[i] = wbuf(chans[i], chans[i], 3, 3);
    wtS[i] = hassc[i] ? wbuf(chans[i], cins[i], 1, 1) : nullptr;
  }

  // ---- weight transforms (OIHW f32 -> [O][K] bf16) ----
  auto prep = [&](const float* s, bf16* d, int O, int I, int KH, int KW) {
    int n = O * I * KH * KW;
    prep_w<<<(n + 255) / 256, 256, 0, stream>>>(s, d, O, I, KH, KW);
  };
  prep(w0, wt0, 64, 3, 3, 3);
  for (int i = 0; i < 8; ++i) {
    prep(blk[i].w1, wt1[i], chans[i], cins[i], 3, 3);
    prep(blk[i].w2, wt2[i], chans[i], chans[i], 3, 3);
    if (hassc[i]) prep(blk[i].ws, wtS[i], chans[i], cins[i], 1, 1);
  }

  auto ilog2 = [](int v) { int s = 0; while ((1 << s) < v) ++s; return s; };

  auto conv = [&](const bf16* in, const bf16* wt, int Hin, int Cin, int Cout,
                  int s, int pad, int Ksz) -> int {
    int Ho = (s == 1) ? Hin : Hin / 2;
    int M = NB * Ho * Ho;
    int wsft = ilog2(Ho);
    dim3 g(M / 64, Cout / 64);
    conv_wmma<<<g, 64, 0, stream>>>(in, wt, convF, Hin, Hin, Cin, Cout,
                                    s, pad, Ksz, Ksz, wsft, 2 * wsft);
    return M;
  };

  auto bn = [&](int M, int C, const float* g_, const float* b_, int relu,
                const bf16* resid, int add, bf16* outp) {
    zero_f32<<<4, 256, 0, stream>>>(stats, 1024);
    bn_stats<<<dim3(C / 64, 256), 64, 0, stream>>>(convF, stats, stats + 512, M, C);
    int n = M * C;
    bn_apply<<<(n + 255) / 256, 256, 0, stream>>>(convF, stats, stats + 512,
                                                  g_, b_, resid, outp,
                                                  n, C - 1, M, relu, add);
  };

  // ---- prologue: x -> NHWC bf16, conv1 + BN + relu ----
  {
    int total = NB * 3 * 1024;
    nchw_to_nhwc<<<(total + 255) / 256, 256, 0, stream>>>(x_in, actB, 3, 1024, total);
    int M = conv(actB, wt0, 32, 3, 64, 1, 1, 3);
    bn(M, 64, g0, b0, /*relu=*/1, actS, /*add=*/0, actA);
  }

  // ---- 8 basic blocks; block input/output lives in actA ----
  for (int i = 0; i < 8; ++i) {
    const int Hin = hins[i], Ci = cins[i], Co = chans[i], s = strides[i];
    const int Ho = (s == 1) ? Hin : Hin / 2;
    const int M = NB * Ho * Ho;

    // shortcut -> actS (already ReLU'd)
    if (hassc[i]) {
      conv(actA, wtS[i], Hin, Ci, Co, s, 0, 1);
      bn(M, Co, blk[i].gs, blk[i].bs, 1, actS, 0, actS);
    } else {
      int n = M * Co;
      relu_bf16_k<<<(n + 255) / 256, 256, 0, stream>>>(actA, actS, n);
    }
    // main path
    conv(actA, wt1[i], Hin, Ci, Co, s, 1, 3);
    bn(M, Co, blk[i].g1, blk[i].b1, 1, actS, 0, actB);
    conv(actB, wt2[i], Ho, Co, Co, 1, 1, 3);
    bn(M, Co, blk[i].g2, blk[i].b2, 0, actS, 1, actA);
  }

  // ---- avgpool + FC ----
  pool_fc<<<NB, 128, 0, stream>>>(actA, Wfc, bias, (float*)d_out);
}
